// TMPNN_26843545600522
// MI455X (gfx1250) — matrix operand order repaired and checked by reading, not compiled, not run on hardware
//
#include <hip/hip_runtime.h>
#include <stdint.h>

// ---------------------------------------------------------------------------
// TMPNN Taylor-map, 7 residual steps of X += [1, X, X(x)X] @ W  (B=65536,d=64)
//
// Quad term per 16-row tile:  sum_i (x_i * Xrow) @ Wq[i]
//   -> per-lane row scaling of the resident A fragment (v_pk_mul_f16),
//      f16 inputs / f32 accumulate via v_wmma_f32_16x16x32_f16.
// W delivery: Tensor Data Mover streams i-major Wq chunks (32 KB) into a
//   double-buffered LDS stage (pad feature gives a 144B row pitch -> 16
//   distinct banks per B-fragment read). TENSORcnt + workgroup barriers.
// TDM issue is guarded by a *scalar* (readfirstlane) branch: TENSOR ops
// ignore EXEC, so the guard must be a scalar skip, not an EXEC mask.
//
// Workspace: [0)        Wq2 f16  [i=64][n=64][j=64]      = 524288 B
//            [524288)   WlinT f16 [n=64][k=64]           =   8192 B
//            [532480)   bias f32 [64]                    =    256 B
//            [540672)   X ping buffer f32, 65536 x 64    = 16 MiB
// ---------------------------------------------------------------------------

typedef _Float16 v16h __attribute__((ext_vector_type(16)));
typedef _Float16 v8h  __attribute__((ext_vector_type(8)));
typedef _Float16 v4h  __attribute__((ext_vector_type(4)));
typedef float    v8f  __attribute__((ext_vector_type(8)));
typedef float    v4f  __attribute__((ext_vector_type(4)));
typedef unsigned int u32x4 __attribute__((ext_vector_type(4)));
typedef int          i32x4 __attribute__((ext_vector_type(4)));
typedef int          i32x8 __attribute__((ext_vector_type(8)));

union Frag16 { v16h v; v8h h[2]; };
union Acc    { v8f  v; float f[8]; };

#define DD             64
#define THREADS        256            // 8 waves x 32 rows
#define ROWS_PER_BLOCK 256
#define XH_PITCH       80             // halfs (160B rows, 16B aligned)
#define CHUNK_I        4              // i's per TDM chunk
#define NCHUNK         (DD / CHUNK_I) // 16
#define ROW_PITCH_H    72             // halfs per n-row in LDS (64 + 8 pad)
#define I_PITCH_H      (64 * ROW_PITCH_H)          // 4608 halfs per i
#define CHUNK_H        (CHUNK_I * I_PITCH_H)       // 18432 halfs (36864 B)
#define CHUNK_DW       (CHUNK_I * 64 * 32)         // 8192 DWORDs moved per chunk

// --- W convert:  W (4161 x 64 f32, row-major) -> i-major f16 ---------------
__global__ void tmpnn_convert_w(const float* __restrict__ W,
                                _Float16* __restrict__ Wq2,    // [64][64][64]
                                _Float16* __restrict__ WlinT,  // [64][64]
                                float* __restrict__ biasF)     // [64]
{
    int idx = blockIdx.x * blockDim.x + threadIdx.x;   // 0 .. 64*4096-1
    int i = idx >> 12;
    int r = idx & 4095;
    int n = r >> 6;
    int j = r & 63;
    Wq2[idx] = (_Float16)W[(size_t)(65 + i * DD + j) * DD + n];
    if (idx < 4096) {
        int n2 = idx >> 6, k = idx & 63;
        WlinT[n2 * DD + k] = (_Float16)W[(size_t)(1 + k) * DD + n2];
    }
    if (idx < DD) biasF[idx] = W[idx];                 // W[0][n]
}

// Issue one TDM chunk copy: 32 KB contiguous global -> padded LDS tile.
__device__ __forceinline__ void tdm_issue(const _Float16* gsrc, _Float16* ldst)
{
#if __has_builtin(__builtin_amdgcn_tensor_load_to_lds)
    uint64_t ga = (uint64_t)(uintptr_t)gsrc;
    uint32_t la = (uint32_t)(uintptr_t)ldst;  // low 32 bits of generic = LDS byte offset
    u32x4 g0;
    g0.x = 1u;                                // count=1 (valid user descriptor)
    g0.y = la;                                // lds_addr
    g0.z = (uint32_t)ga;                      // global_addr[31:0]
    g0.w = (uint32_t)(ga >> 32) | (2u << 30); // global_addr[56:32] | type=2 (image)
    i32x8 g1;
    g1[0] = (2 << 16)        // data_size = 4B (dims in DWORDs)
          | (1 << 20)        // pad_enable
          | (4 << 22)        // pad_interval: every 32 DWORDs
          | (3 << 25);       // pad_amount: 4 DWORDs -> 144B row pitch in LDS
    g1[1] = (int)((unsigned)CHUNK_DW << 16);  // tensor_dim0[15:0]  (8192 DW)
    g1[2] = 0;                                // tensor_dim0 hi / tensor_dim1 lo
    g1[3] = (int)((unsigned)CHUNK_DW << 16);  // tile_dim0 = 8192 DW (1-D tile)
    g1[4] = 0; g1[5] = 0; g1[6] = 0; g1[7] = 0;
    i32x4 z4 = {0, 0, 0, 0};
#if defined(__clang_major__) && (__clang_major__ >= 23)
    i32x8 z8 = {0, 0, 0, 0, 0, 0, 0, 0};
    __builtin_amdgcn_tensor_load_to_lds(g0, g1, z4, z4, z8, 0);
#else
    __builtin_amdgcn_tensor_load_to_lds(g0, g1, z4, z4, 0);
#endif
#endif
}

// --- one Taylor step --------------------------------------------------------
__global__ void __launch_bounds__(THREADS)
tmpnn_taylor_step(const float* __restrict__ Xin,
                  const _Float16* __restrict__ Wq2,
                  const _Float16* __restrict__ WlinT,
                  const float* __restrict__ biasF,
                  float* __restrict__ Xout)
{
    __shared__ __attribute__((aligned(16))) _Float16 Xh[ROWS_PER_BLOCK * XH_PITCH];
    __shared__ __attribute__((aligned(16))) _Float16 Wb[2][CHUNK_H];

    const int tid       = threadIdx.x;
    const int lane      = tid & 31;
    // Scalarized wave id: uniform within a wave by construction; readfirstlane
    // pins it in an SGPR so `wave == 0` compiles to a scalar branch (waves 1-7
    // jump over the TDM issue entirely -- TENSOR ops ignore EXEC).
    const int wave      = __builtin_amdgcn_readfirstlane(tid >> 5);
    const int blockRow0 = blockIdx.x * ROWS_PER_BLOCK;

    // ---- stage 256 rows of X into LDS as f16 (A-fragment + scalar feed) ----
    {
        const float* src  = Xin + (size_t)(blockRow0 + tid) * DD;
        _Float16*    dstH = &Xh[tid * XH_PITCH];
        #pragma unroll
        for (int j = 0; j < DD; j += 4) {
            v4f x = *(const v4f*)(src + j);
            v4h hx = { (_Float16)x.x, (_Float16)x.y, (_Float16)x.z, (_Float16)x.w };
            *(v4h*)(dstH + j) = hx;
        }
    }

    // wave 0 kicks off the first W chunk while X staging settles
    if (wave == 0) tdm_issue(Wq2, &Wb[0][0]);
    __syncthreads();

    const int  waveRow0 = wave * 32;          // scalar: 0..224
    const int  mrow     = lane & 15;
    const bool loHalf   = (lane < 16);
    const int  kh       = loHalf ? 0 : 16;    // A-frag half offset (halfs)
    const int  bh       = loHalf ? 0 : 16;    // B-frag half offset (halfs)
    const int  ncol     = lane & 15;

    // ---- resident A fragments (2 mtiles x 2 kchunks) ----
    Frag16 xa[2][2];
    #pragma unroll
    for (int mt = 0; mt < 2; ++mt) {
        const _Float16* rp = &Xh[(waveRow0 + mt * 16 + mrow) * XH_PITCH];
        #pragma unroll
        for (int kc = 0; kc < 2; ++kc) {
            xa[mt][kc].h[0] = *(const v8h*)(rp + kc * 32 + kh);
            xa[mt][kc].h[1] = *(const v8h*)(rp + kc * 32 + 8 + kh);
        }
    }

    Acc acc[2][4];
    #pragma unroll
    for (int mt = 0; mt < 2; ++mt)
        #pragma unroll
        for (int nc = 0; nc < 4; ++nc)
            #pragma unroll
            for (int v = 0; v < 8; ++v) acc[mt][nc].f[v] = 0.0f;

    // ---- linear term: X @ Wlin (B direct from L2; one-shot) ----
    #pragma unroll
    for (int kc = 0; kc < 2; ++kc) {
        #pragma unroll
        for (int nc = 0; nc < 4; ++nc) {
            const _Float16* bp = WlinT + (size_t)(nc * 16 + ncol) * DD + kc * 32 + bh;
            Frag16 b;
            b.h[0] = *(const v8h*)(bp);
            b.h[1] = *(const v8h*)(bp + 8);
            acc[0][nc].v = __builtin_amdgcn_wmma_f32_16x16x32_f16(
                false, xa[0][kc].v, false, b.v, (short)0, acc[0][nc].v, false, false);
            acc[1][nc].v = __builtin_amdgcn_wmma_f32_16x16x32_f16(
                false, xa[1][kc].v, false, b.v, (short)0, acc[1][nc].v, false, false);
        }
    }

    // ---- quadratic term, TDM double-buffered over 16 chunks of 4 i's ----
    for (int c = 0; c < NCHUNK; ++c) {
        if (wave == 0) {                      // scalar branch
            if (c + 1 < NCHUNK) {
                tdm_issue(Wq2 + (size_t)(c + 1) * (CHUNK_I * 4096), &Wb[(c + 1) & 1][0]);
#if __has_builtin(__builtin_amdgcn_s_wait_tensorcnt)
                __builtin_amdgcn_s_wait_tensorcnt((short)1);  // chunk c done (in-order)
#endif
            } else {
#if __has_builtin(__builtin_amdgcn_s_wait_tensorcnt)
                __builtin_amdgcn_s_wait_tensorcnt((short)0);  // drain last chunk
#endif
            }
        }
        __syncthreads();                      // chunk c visible to all 8 waves

        const _Float16* wb = &Wb[c & 1][0];
        #pragma unroll
        for (int il = 0; il < CHUNK_I; ++il) {
            const int i = c * CHUNK_I + il;
            _Float16 s0 = Xh[(waveRow0 + mrow) * XH_PITCH + i];
            _Float16 s1 = Xh[(waveRow0 + 16 + mrow) * XH_PITCH + i];
            const _Float16* bb = wb + il * I_PITCH_H;
            #pragma unroll
            for (int kc = 0; kc < 2; ++kc) {
                v16h a0 = xa[0][kc].v * s0;   // v_pk_mul_f16 row scaling
                v16h a1 = xa[1][kc].v * s1;
                #pragma unroll
                for (int nc = 0; nc < 4; ++nc) {
                    const _Float16* bp =
                        bb + (nc * 16 + ncol) * ROW_PITCH_H + kc * 32 + bh;
                    Frag16 b;
                    b.h[0] = *(const v8h*)(bp);
                    b.h[1] = *(const v8h*)(bp + 8);
                    acc[0][nc].v = __builtin_amdgcn_wmma_f32_16x16x32_f16(
                        false, a0, false, b.v, (short)0, acc[0][nc].v, false, false);
                    acc[1][nc].v = __builtin_amdgcn_wmma_f32_16x16x32_f16(
                        false, a1, false, b.v, (short)0, acc[1][nc].v, false, false);
                }
            }
        }
        __syncthreads();                      // all readers done before c+2 overwrite
    }

    // ---- epilogue: Y = X(f32, re-read from L2-hot global) + bias + acc ----
    float bn[4];
    #pragma unroll
    for (int nc = 0; nc < 4; ++nc) bn[nc] = biasF[nc * 16 + ncol];

    #pragma unroll
    for (int mt = 0; mt < 2; ++mt) {
        #pragma unroll
        for (int nc = 0; nc < 4; ++nc) {
            const int n = nc * 16 + ncol;
            #pragma unroll
            for (int v = 0; v < 8; ++v) {
                int m = loHalf ? v : (v + 8);          // C/D layout
                int r = waveRow0 + mt * 16 + m;
                size_t gi = (size_t)(blockRow0 + r) * DD + n;
                Xout[gi] = Xin[gi] + bn[nc] + acc[mt][nc].f[v];
            }
        }
    }
}

extern "C" void kernel_launch(void* const* d_in, const int* in_sizes, int n_in,
                              void* d_out, int out_size, void* d_ws, size_t ws_size,
                              hipStream_t stream) {
    const float* x = (const float*)d_in[0];   // (B, 64) f32
    const float* W = (const float*)d_in[1];   // (4161, 64) f32
    float* out = (float*)d_out;

    char* ws = (char*)d_ws;
    _Float16* Wq2   = (_Float16*)(ws);             // 512 KB
    _Float16* WlinT = (_Float16*)(ws + 524288);    // 8 KB
    float*    biasF = (float*)(ws + 532480);       // 256 B
    float*    xtmp  = (float*)(ws + 540672);       // 16 MiB ping buffer

    const int B = in_sizes[0] / DD;                // 65536

    tmpnn_convert_w<<<(64 * 4096) / 256, 256, 0, stream>>>(W, Wq2, WlinT, biasF);

    // 7 residual steps; ping-pong ws <-> d_out so step 7 lands in d_out.
    dim3 grid(B / ROWS_PER_BLOCK), block(THREADS);
    const float* cur = x;
    for (int s = 0; s < 7; ++s) {
        float* dst = (s & 1) ? xtmp : out;   // s = 0,2,4,6 -> out; final s=6 -> out
        tmpnn_taylor_step<<<grid, block, 0, stream>>>(cur, Wq2, WlinT, biasF, dst);
        cur = dst;
    }
}
